// Decoder_78778290143866
// MI455X (gfx1250) — matrix operand order repaired and checked by reading, not compiled, not run on hardware
//
#include <hip/hip_runtime.h>
#include <math.h>

// ---------------------------------------------------------------------------
// Dims (fixed by the reference)
// ---------------------------------------------------------------------------
#define Bb 32
#define Ss 100
#define Ii 300
#define Hh 300
#define H2 600
#define Ee 300
#define Dd 30000
#define Vv 50000
#define Tt 30
#define CATW 1500          // H2 + E + H2
#define CATWP 1504         // CATW rounded up to K-chunk multiple (32)
#define NROWS ((Tt - 1) * Bb)   // 928
#define NEGV  (-10000000.0f)

typedef __attribute__((ext_vector_type(16))) __bf16 v16bf;
typedef __attribute__((ext_vector_type(8)))  float  v8f;
typedef int v4i_vs __attribute__((__vector_size__(16)));   // matches builtin param

union BfPack { uint4 u; __bf16 h[8]; };

__device__ __forceinline__ float sigmoidf_(float x) { return 1.0f / (1.0f + expf(-x)); }

// CDNA5 async global->LDS copy availability (device pass only)
#if defined(__gfx1250__) && __has_builtin(__builtin_amdgcn_global_load_async_to_lds_b128)
#define ASYNC_LDS 1
#else
#define ASYNC_LDS 0
#endif

__device__ __forceinline__ void copy16_g2l(const __bf16* gptr, __bf16* lptr)
{
#if ASYNC_LDS
  // Builtin signature (from hipcc diagnostic): param0 is non-const
  // 'int __vector_size__(16) AS(1)*'. Strip const / re-pointee with a generic
  // cast, then addrspace-cast.
  v4i_vs* g_gen = (v4i_vs*)gptr;
  v4i_vs* l_gen = (v4i_vs*)lptr;
  __builtin_amdgcn_global_load_async_to_lds_b128(
      (__attribute__((address_space(1))) v4i_vs*)g_gen,
      (__attribute__((address_space(3))) v4i_vs*)l_gen, 0, 0);
#else
  *reinterpret_cast<uint4*>(lptr) = *reinterpret_cast<const uint4*>(gptr);
#endif
}

__device__ __forceinline__ void wait_async_lds()
{
#if ASYNC_LDS
#if __has_builtin(__builtin_amdgcn_s_wait_asynccnt)
  __builtin_amdgcn_s_wait_asynccnt(0);
#else
  asm volatile("s_wait_asynccnt 0x0" ::: "memory");
#endif
#endif
}

// ---------------------------------------------------------------------------
// Generic WMMA bf16 GEMM:  C[row_map[m]] = act(A@W^T + bias + Cinit) * row_scale
//   A: [M,K] fp32 row-major (stride lda).
//   WBF=false: W [N,K] fp32 row-major; converted to bf16 during staging.
//   WBF=true : W [N,ldw] bf16 row-major, K-padded with zeros to a multiple of
//              32 (ldw). Staged with async global->LDS b128 copies (ASYNCcnt)
//              when the toolchain exposes the builtin.
//   Block tile 64(M)x128(N), K-chunk 32 (== WMMA K). 8 waves in a 2x4 grid;
//   each wave owns a 32x32 macro tile = 2x2 WMMA tiles -> 4 v_wmma per chunk.
//   Steady-state staging clamps the source ROW instead of branching: OOB rows
//   only feed discarded C rows/cols (a matmul never mixes rows).
//   LDS row stride 40 bf16 (80B) => 16B-aligned, conflict-free b128 accesses.
// Fragment layout as staged (wave32, contiguous k, same for A and B):
//   A 16x32 bf16: lane L: m=L%16, half=L/16; elem j -> K=half*16+j
//   B 32x16 bf16: lane L: n=L%16, half=L/16; elem j -> K=half*16+j
//   C/D 16x16 f32: elem i -> M = half*8+i, N = L%16
// ---------------------------------------------------------------------------
template <bool WBF>
__global__ __launch_bounds__(256) void gemm_kernel(
    const float* __restrict__ A, int lda,
    const void* __restrict__ Wv, int ldw,
    const float* __restrict__ bias,
    const float* __restrict__ Cinit, int ldci,
    float* __restrict__ C, int ldc,
    const int* __restrict__ row_map,
    const float* __restrict__ row_scale, int rs_stride,
    int M, int N, int K, int act)
{
  __shared__ __attribute__((aligned(16))) __bf16 As[64][40];
  __shared__ __attribute__((aligned(16))) __bf16 Bs[128][40];

  const int tid  = threadIdx.x;
  const int lane = tid & 31;
  const int wave = tid >> 5;
  const int wm   = wave >> 2;        // 0..1  (M half)
  const int wn   = wave & 3;         // 0..3  (N quarter)
  const int m0   = blockIdx.x * 64;
  const int n0   = blockIdx.y * 128;

  v8f acc[2][2] = {};

  const int rowL = tid >> 2;         // 0..63
  const int col0 = (tid & 3) * 8;    // 0,8,16,24
  const int gmc  = min(m0 + rowL, M - 1);          // clamped A row
  const int gn0c = min(n0 + rowL, N - 1);          // clamped W row (lower)
  const int gn1c = min(n0 + 64 + rowL, N - 1);     // clamped W row (upper)

  const float*  Wf = (const float*)Wv;
  const __bf16* Wb = (const __bf16*)Wv;

  for (int k0 = 0; k0 < K; k0 += 32) {
    const int gk = k0 + col0;
    // ---------------- stage A (fp32 -> bf16, packed b128 commit) ----------
    if (gk + 7 < K) {
      const float4* p = reinterpret_cast<const float4*>(A + (size_t)gmc * lda + gk);
      float4 f0 = p[0], f1 = p[1];
      BfPack pk;
      pk.h[0]=(__bf16)f0.x; pk.h[1]=(__bf16)f0.y; pk.h[2]=(__bf16)f0.z; pk.h[3]=(__bf16)f0.w;
      pk.h[4]=(__bf16)f1.x; pk.h[5]=(__bf16)f1.y; pk.h[6]=(__bf16)f1.z; pk.h[7]=(__bf16)f1.w;
      *reinterpret_cast<uint4*>(&As[rowL][col0]) = pk.u;
      if (gk + 39 < K) __builtin_prefetch(A + (size_t)gmc * lda + gk + 32, 0, 1);
    } else {
      BfPack pa;
      #pragma unroll
      for (int i = 0; i < 8; ++i) {
        int kk = gk + i;
        pa.h[i] = (__bf16)((kk < K) ? A[(size_t)gmc * lda + kk] : 0.0f);
      }
      *reinterpret_cast<uint4*>(&As[rowL][col0]) = pa.u;
    }
    // ---------------- stage W ---------------------------------------------
    if (WBF) {
      // bf16 source, zero K-padded: branch-free 16B slots, async when possible
      const int slot0 = tid * 2;     // 512 slots of 16B cover 128x32 bf16
      #pragma unroll
      for (int q = 0; q < 2; ++q) {
        int slot = slot0 + q;
        int wr = slot >> 2;            // 0..127
        int wc = (slot & 3) * 8;       // 0,8,16,24
        int gn = min(n0 + wr, N - 1);
        const __bf16* gp = Wb + (size_t)gn * ldw + k0 + wc;
        copy16_g2l(gp, &Bs[wr][wc]);
        if (k0 + 32 < K) __builtin_prefetch(gp + 32, 0, 1);
      }
    } else if (gk + 7 < K) {
      {
        const float4* p = reinterpret_cast<const float4*>(Wf + (size_t)gn0c * ldw + gk);
        float4 f0 = p[0], f1 = p[1];
        BfPack pk;
        pk.h[0]=(__bf16)f0.x; pk.h[1]=(__bf16)f0.y; pk.h[2]=(__bf16)f0.z; pk.h[3]=(__bf16)f0.w;
        pk.h[4]=(__bf16)f1.x; pk.h[5]=(__bf16)f1.y; pk.h[6]=(__bf16)f1.z; pk.h[7]=(__bf16)f1.w;
        *reinterpret_cast<uint4*>(&Bs[rowL][col0]) = pk.u;
      }
      {
        const float4* p = reinterpret_cast<const float4*>(Wf + (size_t)gn1c * ldw + gk);
        float4 f0 = p[0], f1 = p[1];
        BfPack pk;
        pk.h[0]=(__bf16)f0.x; pk.h[1]=(__bf16)f0.y; pk.h[2]=(__bf16)f0.z; pk.h[3]=(__bf16)f0.w;
        pk.h[4]=(__bf16)f1.x; pk.h[5]=(__bf16)f1.y; pk.h[6]=(__bf16)f1.z; pk.h[7]=(__bf16)f1.w;
        *reinterpret_cast<uint4*>(&Bs[64 + rowL][col0]) = pk.u;
      }
      if (gk + 39 < K) {
        __builtin_prefetch(Wf + (size_t)gn0c * ldw + gk + 32, 0, 1);
        __builtin_prefetch(Wf + (size_t)gn1c * ldw + gk + 32, 0, 1);
      }
    } else {
      BfPack pb0, pb1;
      #pragma unroll
      for (int i = 0; i < 8; ++i) {
        int kk = gk + i;
        bool okk = kk < K;
        pb0.h[i] = (__bf16)(okk ? Wf[(size_t)gn0c * ldw + kk] : 0.0f);
        pb1.h[i] = (__bf16)(okk ? Wf[(size_t)gn1c * ldw + kk] : 0.0f);
      }
      *reinterpret_cast<uint4*>(&Bs[rowL][col0])      = pb0.u;
      *reinterpret_cast<uint4*>(&Bs[64 + rowL][col0]) = pb1.u;
    }
    if (WBF) wait_async_lds();
    __syncthreads();

    const int half = lane >> 4;
    const int lr   = lane & 15;

    v16bf bfrag[2], afrag[2];
    #pragma unroll
    for (int sn = 0; sn < 2; ++sn) {
      BfPack p0, p1;
      p0.u = *reinterpret_cast<const uint4*>(&Bs[wn*32 + sn*16 + lr][half*16]);
      p1.u = *reinterpret_cast<const uint4*>(&Bs[wn*32 + sn*16 + lr][half*16 + 8]);
      #pragma unroll
      for (int j = 0; j < 8; ++j) { bfrag[sn][j] = p0.h[j]; bfrag[sn][8+j] = p1.h[j]; }
    }
    #pragma unroll
    for (int sm = 0; sm < 2; ++sm) {
      BfPack q0, q1;
      q0.u = *reinterpret_cast<const uint4*>(&As[wm*32 + sm*16 + lr][half*16]);
      q1.u = *reinterpret_cast<const uint4*>(&As[wm*32 + sm*16 + lr][half*16 + 8]);
      #pragma unroll
      for (int j = 0; j < 8; ++j) { afrag[sm][j] = q0.h[j]; afrag[sm][8+j] = q1.h[j]; }
    }
    #pragma unroll
    for (int sm = 0; sm < 2; ++sm)
      #pragma unroll
      for (int sn = 0; sn < 2; ++sn)
        acc[sm][sn] = __builtin_amdgcn_wmma_f32_16x16x32_bf16(
            false, afrag[sm], false, bfrag[sn], (short)0, acc[sm][sn], false, false);
    __syncthreads();
  }

  // ---- epilogue: bias + Cinit + activation + row scale + row remap ----
  const int half = lane >> 4;
  const int lr   = lane & 15;
  #pragma unroll
  for (int sn = 0; sn < 2; ++sn) {
    const int gn = n0 + wn*32 + sn*16 + lr;
    if (gn < N) {
      const float bv = bias ? bias[gn] : 0.0f;
      #pragma unroll
      for (int sm = 0; sm < 2; ++sm) {
        #pragma unroll
        for (int i = 0; i < 8; ++i) {
          int gm = m0 + wm*32 + sm*16 + half*8 + i;
          if (gm < M) {
            float v = acc[sm][sn][i] + bv;
            if (Cinit)     v += Cinit[(size_t)gm * ldci + gn];
            if (act == 1)  v = tanhf(v);
            if (row_scale) v *= row_scale[(size_t)gm * rs_stride];
            int orow = row_map ? row_map[gm] : gm;
            C[(size_t)orow * ldc + gn] = v;
          }
        }
      }
    }
  }
}

// ---------------------------------------------------------------------------
// One-shot fp32 -> bf16 conversion of Wout, K-padded [Dd][CATWP] with zeros so
// the bf16-W GEMM staging is branch-free and 16B aligned everywhere.
// ---------------------------------------------------------------------------
__global__ __launch_bounds__(256) void convert_wbf_kernel(
    const float* __restrict__ W, __bf16* __restrict__ Wb)
{
  const int slots_per_row = CATWP / 8;               // 188
  int slot = blockIdx.x * 256 + threadIdx.x;
  if (slot >= Dd * slots_per_row) return;
  int row = slot / slots_per_row;
  int c8  = (slot % slots_per_row) * 8;
  BfPack pk;
  if (c8 + 7 < CATW) {
    const float4* p = reinterpret_cast<const float4*>(W + (size_t)row * CATW + c8);
    float4 f0 = p[0], f1 = p[1];
    pk.h[0]=(__bf16)f0.x; pk.h[1]=(__bf16)f0.y; pk.h[2]=(__bf16)f0.z; pk.h[3]=(__bf16)f0.w;
    pk.h[4]=(__bf16)f1.x; pk.h[5]=(__bf16)f1.y; pk.h[6]=(__bf16)f1.z; pk.h[7]=(__bf16)f1.w;
  } else {
    #pragma unroll
    for (int i = 0; i < 8; ++i)
      pk.h[i] = (__bf16)((c8 + i < CATW) ? W[(size_t)row * CATW + c8 + i] : 0.0f);
  }
  *reinterpret_cast<uint4*>(Wb + (size_t)row * CATWP + c8) = pk.u;
}

// ---------------------------------------------------------------------------
// Bi-GRU encoder recurrence (one workgroup per (batch row, direction)).
// ---------------------------------------------------------------------------
__global__ __launch_bounds__(256) void encoder_rec_kernel(
    const float* __restrict__ gi_f, const float* __restrict__ gi_b,
    const float* __restrict__ Whh_f, const float* __restrict__ bhh_f,
    const float* __restrict__ Whh_b, const float* __restrict__ bhh_b,
    float* __restrict__ enc, float* __restrict__ hcat)
{
  const int b   = blockIdx.x & 31;
  const int dir = blockIdx.x >> 5;
  const float* gi  = dir ? gi_b  : gi_f;
  const float* Whh = dir ? Whh_b : Whh_f;
  const float* bhh = dir ? bhh_b : bhh_f;

  __shared__ __attribute__((aligned(16))) float h[304];
  __shared__ float gh[900];
  const int tid = threadIdx.x;

  for (int j = tid; j < 300; j += 256) h[j] = 0.0f;
  __syncthreads();

  for (int step = 0; step < Ss; ++step) {
    const int s = dir ? (Ss - 1 - step) : step;
    for (int j = tid; j < 900; j += 256) {
      const float4* wr = reinterpret_cast<const float4*>(Whh + (size_t)j * 300);
      float acc = 0.0f;
      #pragma unroll 5
      for (int k4 = 0; k4 < 75; ++k4) {
        float4 w  = wr[k4];
        float4 hv = *reinterpret_cast<const float4*>(&h[k4 * 4]);
        acc += w.x*hv.x + w.y*hv.y + w.z*hv.z + w.w*hv.w;
      }
      gh[j] = acc + bhh[j];
    }
    __syncthreads();
    const float* gir = gi + ((size_t)b * Ss + s) * 900;
    for (int j = tid; j < 300; j += 256) {
      float r  = sigmoidf_(gir[j]       + gh[j]);
      float z  = sigmoidf_(gir[300 + j] + gh[300 + j]);
      float n  = tanhf    (gir[600 + j] + r * gh[600 + j]);
      float h2 = (1.0f - z) * n + z * h[j];
      h[j] = h2;
      enc[((size_t)b * Ss + s) * H2 + dir * Hh + j] = h2;
    }
    __syncthreads();
  }
  for (int j = tid; j < 300; j += 256)
    hcat[(size_t)b * H2 + dir * Hh + j] = h[j];
}

// ---------------------------------------------------------------------------
// Fused decoder step (one block per batch row):
//   phase 1 (t>0): GRU pointwise for step t-1 -> h, store cat_all[t-1]
//   phase 2 (t<29): q = h@Wq^T+bq; masked additive attention; att context
// ---------------------------------------------------------------------------
__global__ __launch_bounds__(256) void dec_step_kernel(
    int t,
    const float* __restrict__ keyed, const float* __restrict__ enc,
    const int*   __restrict__ mask,
    const float* __restrict__ Wq, const float* __restrict__ bq,
    const float* __restrict__ We, const float* __restrict__ be,
    const float* __restrict__ gi_step, const float* __restrict__ gh_step,
    const float* __restrict__ emb_prev,
    float* __restrict__ hcur, float* __restrict__ cat_all,
    float* __restrict__ att_step, float* __restrict__ attn_all)
{
  const int b = blockIdx.x;
  const int tid = threadIdx.x, lane = tid & 31, wave = tid >> 5;
  __shared__ __attribute__((aligned(16))) float hs[600];
  __shared__ float qs[600];
  __shared__ float es[128];
  __shared__ float redv;

  if (t > 0) {
    const int r = (t - 1) * Bb + b;
    const float* gi = gi_step + (size_t)b * 1800;
    const float* gh = gh_step + (size_t)b * 1800;
    for (int j = tid; j < H2; j += 256) {
      float rr = sigmoidf_(gi[j]        + gh[j]);
      float zz = sigmoidf_(gi[600 + j]  + gh[600 + j]);
      float nn = tanhf    (gi[1200 + j] + rr * gh[1200 + j]);
      float h2 = (1.0f - zz) * nn + zz * hcur[(size_t)b * H2 + j];
      hcur[(size_t)b * H2 + j] = h2;
      cat_all[(size_t)r * CATW + j] = h2;
    }
    for (int j = tid; j < Ee; j += 256)
      cat_all[(size_t)r * CATW + H2 + j] = emb_prev[(size_t)r * Ee + j];
    for (int j = tid; j < H2; j += 256)
      cat_all[(size_t)r * CATW + H2 + Ee + j] = att_step[(size_t)b * H2 + j];
  }
  __syncthreads();
  if (t >= Tt - 1) return;   // finalize-only pass

  for (int j = tid; j < H2; j += 256) hs[j] = hcur[(size_t)b * H2 + j];
  __syncthreads();

  // q = h @ Wq^T + bq
  for (int j = tid; j < H2; j += 256) {
    const float4* wr = reinterpret_cast<const float4*>(Wq + (size_t)j * H2);
    float acc = 0.0f;
    #pragma unroll 5
    for (int k4 = 0; k4 < 150; ++k4) {
      float4 w  = wr[k4];
      float4 hv = *reinterpret_cast<const float4*>(&hs[k4 * 4]);
      acc += w.x*hv.x + w.y*hv.y + w.z*hv.z + w.w*hv.w;
    }
    qs[j] = acc + bq[j];
  }
  __syncthreads();

  // e_s = tanh(keyed + q) . We + be, masked
  for (int s = wave; s < Ss; s += 8) {
    const float* kr = keyed + ((size_t)b * Ss + s) * H2;
    float p = 0.0f;
    for (int j = lane; j < H2; j += 32) p += tanhf(kr[j] + qs[j]) * We[j];
    for (int off = 16; off; off >>= 1) p += __shfl_down(p, off, 32);
    if (lane == 0) es[s] = mask[b * Ss + s] ? (p + be[0]) : NEGV;
  }
  __syncthreads();

  // softmax over S (wave 0)
  if (wave == 0) {
    float m = -3.4e38f;
    for (int s = lane; s < Ss; s += 32) m = fmaxf(m, es[s]);
    for (int off = 16; off; off >>= 1) m = fmaxf(m, __shfl_xor(m, off, 32));
    float sum = 0.0f;
    for (int s = lane; s < Ss; s += 32) { float ex = expf(es[s] - m); es[s] = ex; sum += ex; }
    for (int off = 16; off; off >>= 1) sum += __shfl_xor(sum, off, 32);
    if (lane == 0) redv = 1.0f / sum;
  }
  __syncthreads();
  const float inv = redv;
  for (int s = tid; s < Ss; s += 256) {
    float a = es[s] * inv;
    es[s] = a;
    attn_all[((size_t)t * Bb + b) * Ss + s] = a;
  }
  __syncthreads();

  // att = attn @ enc
  for (int j = tid; j < H2; j += 256) {
    float acc = 0.0f;
    for (int s = 0; s < Ss; ++s) acc += es[s] * enc[((size_t)b * Ss + s) * H2 + j];
    att_step[(size_t)b * H2 + j] = acc;
  }
}

// ---------------------------------------------------------------------------
// Small helper kernels
// ---------------------------------------------------------------------------
__global__ void prep_kernel(const float* __restrict__ emb,
                            const int* __restrict__ true_answer,
                            float* __restrict__ emb_prev,
                            int* __restrict__ row_map)
{
  const int r = blockIdx.x;           // 0..927 ; r = t*32 + b
  const int t = r >> 5, b = r & 31;
  const int tok = true_answer[b * Tt + t];
  for (int j = threadIdx.x; j < Ee; j += 64)
    emb_prev[(size_t)r * Ee + j] = emb[(size_t)tok * Ee + j];
  if (threadIdx.x == 0) row_map[r] = b * Tt + (t + 1);
}

__global__ void gate_kernel(const float* __restrict__ gcat, float* __restrict__ g)
{
  const int r = blockIdx.x * 256 + threadIdx.x;
  if (r < NROWS) {
    float s0 = sigmoidf_(gcat[r * 2 + 0]);
    float s1 = sigmoidf_(gcat[r * 2 + 1]);
    float m  = fmaxf(s0, s1);
    float e0 = expf(s0 - m), e1 = expf(s1 - m);
    float inv = 1.0f / (e0 + e1);
    g[r * 2 + 0] = e0 * inv;
    g[r * 2 + 1] = e1 * inv;
  }
}

__global__ void zero_first_kernel(float* __restrict__ out)
{
  const int idx = blockIdx.x * 256 + threadIdx.x;
  if (idx < Bb * Dd) {
    int b = idx / Dd, n = idx % Dd;
    out[((size_t)b * Tt) * Dd + n] = 0.0f;   // results[:,0,:] = 0
  }
}

__global__ void blend_kernel(const float* __restrict__ g,
                             const float* __restrict__ attn_all,
                             const int* __restrict__ span_idx,
                             float* __restrict__ out)
{
  const int r = blockIdx.x;            // t*32 + b
  const int t = r >> 5, b = r & 31;
  const float g1 = g[r * 2 + 1];
  const size_t base = ((size_t)b * Tt + t + 1) * Dd;
  for (int s = threadIdx.x; s < Ss; s += 128) {
    int idx = span_idx[b * Ss + s];    // unique per row -> no collisions
    if (idx < Dd) out[base + idx] += g1 * attn_all[(size_t)r * Ss + s];
  }
}

// ---------------------------------------------------------------------------
// Host side
// ---------------------------------------------------------------------------
static inline void launch_gemm(hipStream_t st,
    const float* A, int lda, const float* W, int ldw, const float* bias,
    const float* Cinit, int ldci, float* C, int ldc,
    const int* rm, const float* rs, int rss,
    int M, int N, int K, int act)
{
  dim3 grid((M + 63) / 64, (N + 127) / 128);
  gemm_kernel<false><<<grid, 256, 0, st>>>(A, lda, (const void*)W, ldw, bias,
                                           Cinit, ldci, C, ldc, rm, rs, rss,
                                           M, N, K, act);
}

static inline void launch_gemm_wbf(hipStream_t st,
    const float* A, int lda, const __bf16* W, int ldw, const float* bias,
    float* C, int ldc, const int* rm, const float* rs, int rss,
    int M, int N, int K)
{
  dim3 grid((M + 63) / 64, (N + 127) / 128);
  gemm_kernel<true><<<grid, 256, 0, st>>>(A, lda, (const void*)W, ldw, bias,
                                          nullptr, 0, C, ldc, rm, rs, rss,
                                          M, N, K, 0);
}

extern "C" void kernel_launch(void* const* d_in, const int* in_sizes, int n_in,
                              void* d_out, int out_size, void* d_ws, size_t ws_size,
                              hipStream_t stream) {
  const float* span   = (const float*)d_in[0];
  const int*   t_ans  = (const int*)  d_in[1];
  const int*   mask   = (const int*)  d_in[2];
  const int*   sp_idx = (const int*)  d_in[3];
  const float* emb    = (const float*)d_in[4];
  const float* eWih_f = (const float*)d_in[5];
  const float* eWhh_f = (const float*)d_in[6];
  const float* ebih_f = (const float*)d_in[7];
  const float* ebhh_f = (const float*)d_in[8];
  const float* eWih_b = (const float*)d_in[9];
  const float* eWhh_b = (const float*)d_in[10];
  const float* ebih_b = (const float*)d_in[11];
  const float* ebhh_b = (const float*)d_in[12];
  const float* fcW    = (const float*)d_in[13];
  const float* fcb    = (const float*)d_in[14];
  const float* Wk     = (const float*)d_in[15];
  const float* bk     = (const float*)d_in[16];
  const float* Wq     = (const float*)d_in[17];
  const float* bq     = (const float*)d_in[18];
  const float* We     = (const float*)d_in[19];
  const float* be     = (const float*)d_in[20];
  const float* cWih   = (const float*)d_in[21];
  const float* cWhh   = (const float*)d_in[22];
  const float* cbih   = (const float*)d_in[23];
  const float* cbhh   = (const float*)d_in[24];
  const float* Wout   = (const float*)d_in[25];
  const float* bout   = (const float*)d_in[26];
  const float* Wg     = (const float*)d_in[27];
  const float* bg     = (const float*)d_in[28];
  float* out = (float*)d_out;

  // Workspace layout (floats)
  float* ws = (float*)d_ws;
  size_t o = 0;
  float* gi_f     = ws + o; o += (size_t)Bb*Ss*900;        // 2,880,000
  float* gi_b     = ws + o; o += (size_t)Bb*Ss*900;
  float* enc      = ws + o; o += (size_t)Bb*Ss*H2;         // 1,920,000
  float* keyed    = ws + o; o += (size_t)Bb*Ss*H2;
  float* hcat     = ws + o; o += (size_t)Bb*H2;
  float* hcur     = ws + o; o += (size_t)Bb*H2;
  float* emb_prev = ws + o; o += (size_t)NROWS*Ee;         // 278,400
  float* gi_pre   = ws + o; o += (size_t)NROWS*1800;       // 1,670,400
  float* gi_step  = ws + o; o += (size_t)Bb*1800;
  float* gh_step  = ws + o; o += (size_t)Bb*1800;
  float* att_step = ws + o; o += (size_t)Bb*H2;
  float* cat_all  = ws + o; o += (size_t)NROWS*CATW;       // 1,392,000
  float* attn_all = ws + o; o += (size_t)NROWS*Ss;
  float* gcat     = ws + o; o += (size_t)NROWS*2;
  float* gbuf     = ws + o; o += (size_t)NROWS*2;
  int*   row_map  = (int*)(ws + o); o += NROWS;
  // optional bf16 copy of Wout (K-padded), used only if ws is large enough
  __bf16* wout_bf = reinterpret_cast<__bf16*>(ws + o);
  const size_t need_bytes = o * sizeof(float) + (size_t)Dd * CATWP * sizeof(__bf16);
  const bool use_bf16w = ws_size >= need_bytes;   // fixed per harness -> deterministic

  // 1) Hoisted encoder input projections: gi = span @ Wih^T + bih  [3200,900]
  launch_gemm(stream, span, Ii, eWih_f, Ii, ebih_f, nullptr, 0,
              gi_f, 900, nullptr, nullptr, 0, Bb*Ss, 900, Ii, 0);
  launch_gemm(stream, span, Ii, eWih_b, Ii, ebih_b, nullptr, 0,
              gi_b, 900, nullptr, nullptr, 0, Bb*Ss, 900, Ii, 0);

  // 2) Teacher-forced embeddings + output row map
  prep_kernel<<<NROWS, 64, 0, stream>>>(emb, t_ans, emb_prev, row_map);

  // 2b) Pre-convert Wout to bf16 (overlaps with recurrence work upstream)
  if (use_bf16w) {
    int slots = Dd * (CATWP / 8);
    convert_wbf_kernel<<<(slots + 255) / 256, 256, 0, stream>>>(Wout, wout_bf);
  }

  // 3) Bi-GRU recurrence: one launch, 64 independent (b,dir) workgroups
  encoder_rec_kernel<<<64, 256, 0, stream>>>(gi_f, gi_b, eWhh_f, ebhh_f,
                                             eWhh_b, ebhh_b, enc, hcat);

  // 4) hidden = tanh([hf,hb] @ fcW^T + fcb) -> initial decoder state
  launch_gemm(stream, hcat, H2, fcW, H2, fcb, nullptr, 0,
              hcur, H2, nullptr, nullptr, 0, Bb, H2, H2, 1);

  // 5) Hoisted attention keys: keyed = enc @ Wk^T + bk  [3200,600]
  launch_gemm(stream, enc, H2, Wk, H2, bk, nullptr, 0,
              keyed, H2, nullptr, nullptr, 0, Bb*Ss, H2, H2, 0);

  // 6) Precompute prev_emb contribution to decoder GRU gates (+cbih) for all t
  launch_gemm(stream, emb_prev, Ee, cWih + H2, H2 + Ee, cbih, nullptr, 0,
              gi_pre, 1800, nullptr, nullptr, 0, NROWS, 1800, Ee, 0);

  // 7) Decoder recurrence: 29 steps of {fused pointwise+attention, gate GEMMs}
  for (int t = 0; t < Tt - 1; ++t) {
    dec_step_kernel<<<Bb, 256, 0, stream>>>(t, keyed, enc, mask, Wq, bq, We, be,
                                            gi_step, gh_step, emb_prev,
                                            hcur, cat_all, att_step, attn_all);
    // gi = gi_pre[t] + att @ cWih[:, :600]^T
    launch_gemm(stream, att_step, H2, cWih, H2 + Ee, nullptr,
                gi_pre + (size_t)t * Bb * 1800, 1800,
                gi_step, 1800, nullptr, nullptr, 0, Bb, 1800, H2, 0);
    // gh = h @ cWhh^T + cbhh
    launch_gemm(stream, hcur, H2, cWhh, H2, cbhh, nullptr, 0,
                gh_step, 1800, nullptr, nullptr, 0, Bb, 1800, H2, 0);
  }
  // finalize last GRU pointwise + cat_all[T-2]
  dec_step_kernel<<<Bb, 256, 0, stream>>>(Tt - 1, keyed, enc, mask, Wq, bq, We, be,
                                          gi_step, gh_step, emb_prev,
                                          hcur, cat_all, att_step, attn_all);

  // 8) Mixture gate: g = softmax(sigmoid(cat @ Wg^T + bg))
  launch_gemm(stream, cat_all, CATW, Wg, CATW, bg, nullptr, 0,
              gcat, 2, nullptr, nullptr, 0, NROWS, 2, CATW, 0);
  gate_kernel<<<(NROWS + 255) / 256, 256, 0, stream>>>(gcat, gbuf);

  // 9) Output: zero t=0 rows, then one big batched vocab GEMM with fused
  //    g0-scale + row scatter into d_out[b, t+1, :]; then sparse copy blend.
  zero_first_kernel<<<(Bb * Dd + 255) / 256, 256, 0, stream>>>(out);
  if (use_bf16w) {
    launch_gemm_wbf(stream, cat_all, CATW, wout_bf, CATWP, bout,
                    out, Dd, row_map, gbuf, 2, NROWS, Dd, CATW);
  } else {
    launch_gemm(stream, cat_all, CATW, Wout, CATW, bout, nullptr, 0,
                out, Dd, row_map, gbuf, 2, NROWS, Dd, CATW, 0);
  }
  blend_kernel<<<NROWS, 128, 0, stream>>>(gbuf, attn_all, sp_idx, out);
}
